// MoEFFN_51505247813859
// MI455X (gfx1250) — compile-verified, matrix operand order checked by reading, not run on hardware
//
#include <hip/hip_runtime.h>
#include <hip/hip_bf16.h>

// ---------------- problem constants ----------------
#define NB 4
#define NT 2048
#define NC 1024
#define NE 8
#define NH 4096
#define NTOK (NB * NT)            // 8192 tokens

// ---------------- WMMA vector types ----------------
typedef __attribute__((ext_vector_type(16))) __bf16         v16bf;
typedef __attribute__((ext_vector_type(8)))  float          v8f;
typedef __attribute__((ext_vector_type(8)))  unsigned short v8u;
typedef __attribute__((ext_vector_type(4)))  int            v4i;

struct bfpair { v8u lo; v8u hi; };

// fp32 -> bf16 via hardware convert
static __device__ __forceinline__ unsigned short f2bf(float f) {
    return __builtin_bit_cast(unsigned short, (__bf16)f);
}

// load a 16x32-bf16 wave fragment: 8 elems at p[0..7], 8 at p[16..23]
static __device__ __forceinline__ v16bf ld_frag(const unsigned short* p) {
    bfpair t;
    t.lo = *(const v8u*)(p);
    t.hi = *(const v8u*)(p + 16);
    return __builtin_bit_cast(v16bf, t);
}

// ---------------- async global->LDS copy (16B per lane) ----------------
#if __has_builtin(__builtin_amdgcn_global_load_async_to_lds_b128)
#define ASYNC_LDS 1
#else
#define ASYNC_LDS 0
#endif

static __device__ __forceinline__ void async_cp16(const unsigned short* g, unsigned short* l) {
#if ASYNC_LDS
    __builtin_amdgcn_global_load_async_to_lds_b128(
        (__attribute__((address_space(1))) v4i*)(void*)g,
        (__attribute__((address_space(3))) v4i*)(void*)l, 0, 0);
#else
    *(uint4*)l = *(const uint4*)g;
#endif
}

template <int N>
static __device__ __forceinline__ void wait_async() {
#if ASYNC_LDS
#if __has_builtin(__builtin_amdgcn_s_wait_asynccnt)
    __builtin_amdgcn_s_wait_asynccnt(N);
#else
    asm volatile("s_wait_asynccnt %0" ::"i"(N) : "memory");
#endif
#endif
}

// ---------------- 0: zero counters ----------------
__global__ void k_init(int* counts, int* cursor) {
    int t = threadIdx.x;
    if (t < NE) { counts[t] = 0; cursor[t] = 0; }
}

// ---------------- 1: router (one wave32 per token) ----------------
__global__ __launch_bounds__(256) void k_router(const float* __restrict__ x,
                                                const float* __restrict__ Wr,
                                                const float* __restrict__ br,
                                                float* __restrict__ probs,
                                                int* __restrict__ eidx,
                                                int* __restrict__ counts) {
    int wave = threadIdx.x >> 5;
    int lane = threadIdx.x & 31;
    int t = blockIdx.x * 8 + wave;                 // grid = NTOK/8 blocks, exact

    const float* xr = x + (size_t)t * NC;
    float acc[NE];
#pragma unroll
    for (int e = 0; e < NE; ++e) acc[e] = 0.0f;

    for (int c = lane; c < NC; c += 32) {
        float xv = xr[c];
        const float4* w4 = (const float4*)(Wr + (size_t)c * NE);
        float4 w0 = w4[0], w1 = w4[1];
        acc[0] += xv * w0.x; acc[1] += xv * w0.y;
        acc[2] += xv * w0.z; acc[3] += xv * w0.w;
        acc[4] += xv * w1.x; acc[5] += xv * w1.y;
        acc[6] += xv * w1.z; acc[7] += xv * w1.w;
    }
#pragma unroll
    for (int e = 0; e < NE; ++e)
        for (int off = 16; off >= 1; off >>= 1)
            acc[e] += __shfl_xor(acc[e], off);

    if (lane == 0) {
        float lg[NE], m = -1e30f;
#pragma unroll
        for (int e = 0; e < NE; ++e) { lg[e] = acc[e] + br[e]; m = fmaxf(m, lg[e]); }
        float s = 0.0f;
#pragma unroll
        for (int e = 0; e < NE; ++e) { lg[e] = __expf(lg[e] - m); s += lg[e]; }
        float inv = 1.0f / s;
        float best = -1.0f; int bi = 0;
#pragma unroll
        for (int e = 0; e < NE; ++e) {
            float p = lg[e] * inv;
            if (p > best) { best = p; bi = e; }
        }
        probs[t] = best;
        eidx[t]  = bi;
        atomicAdd(&counts[bi], 1);
    }
}

// ---------------- 2: fp32 -> bf16 activation convert ----------------
__global__ __launch_bounds__(256) void k_cvt(const float* __restrict__ x,
                                             unsigned short* __restrict__ xb) {
    size_t i = ((size_t)blockIdx.x * 256 + threadIdx.x) * 8;   // grid sized exactly
    float4 a = *(const float4*)(x + i);
    float4 b = *(const float4*)(x + i + 4);
    v8u o;
    o[0] = f2bf(a.x); o[1] = f2bf(a.y); o[2] = f2bf(a.z); o[3] = f2bf(a.w);
    o[4] = f2bf(b.x); o[5] = f2bf(b.y); o[6] = f2bf(b.z); o[7] = f2bf(b.w);
    *(v8u*)(xb + i) = o;
}

// ---------------- 3: weight transpose + convert: fp32 [K][N] -> bf16 [N][K] ----------------
#define WT 64
#define WTS 72   // LDS stride (144B, 16B aligned rows)
__global__ __launch_bounds__(256) void k_wt(const float* __restrict__ src,
                                            unsigned short* __restrict__ dst,
                                            int K, int N) {
    __shared__ __align__(16) unsigned short T[WT * WTS];
    const int e  = blockIdx.z;
    const int n0 = blockIdx.x * WT;
    const int k0 = blockIdx.y * WT;
    const float* s = src + (size_t)e * K * N;
    unsigned short* d = dst + (size_t)e * N * K;

    const int tid = threadIdx.x;
    {   // read 64x64 fp32 tile, store bf16 transposed into LDS
        int r  = tid >> 2;             // 0..63  (k index in tile)
        int c0 = (tid & 3) * 16;       // 0..48
        const float4* sp = (const float4*)(s + (size_t)(k0 + r) * N + n0 + c0);
#pragma unroll
        for (int j = 0; j < 4; ++j) {
            float4 v = sp[j];
            int c = c0 + j * 4;
            T[(c + 0) * WTS + r] = f2bf(v.x);
            T[(c + 1) * WTS + r] = f2bf(v.y);
            T[(c + 2) * WTS + r] = f2bf(v.z);
            T[(c + 3) * WTS + r] = f2bf(v.w);
        }
    }
    __syncthreads();
    {   // write 64x64 bf16 tile rows (now n-major) with b128 stores
        int r  = tid >> 2;             // 0..63  (n index in tile)
        int c0 = (tid & 3) * 16;       // 0..48  (k index)
        uint4 d0 = *(const uint4*)&T[r * WTS + c0];
        uint4 d1 = *(const uint4*)&T[r * WTS + c0 + 8];
        uint4* dp = (uint4*)(d + (size_t)(n0 + r) * K + k0 + c0);
        dp[0] = d0; dp[1] = d1;
    }
}

// ---------------- 4: tiny scan + aux loss ----------------
__global__ void k_scan(const int* __restrict__ counts, int* __restrict__ offs,
                       float* __restrict__ aux_out) {
    if (threadIdx.x == 0 && blockIdx.x == 0) {
        int s = 0;
#pragma unroll
        for (int e = 0; e < NE; ++e) { offs[e] = s; s += counts[e]; }
        offs[NE] = s;
        float tot = (float)(s > 0 ? s : 1);
        float a = 0.0f;
#pragma unroll
        for (int e = 0; e < NE; ++e) {
            float f = (float)counts[e] / tot - (1.0f / NE);
            a += f * f;
        }
        aux_out[0] = a / (float)NE;
    }
}

// ---------------- 5: scatter token ids grouped by expert ----------------
__global__ __launch_bounds__(256) void k_scatter(const int* __restrict__ eidx,
                                                 const int* __restrict__ offs,
                                                 int* __restrict__ cursor,
                                                 int* __restrict__ perm) {
    int t = blockIdx.x * 256 + threadIdx.x;        // grid = NTOK/256, exact
    int e = eidx[t];
    int p = atomicAdd(&cursor[e], 1);
    perm[offs[e] + p] = t;
}

// ---------------- grouped GEMM tiles ----------------
#define TM 128
#define TN 128
#define LSTR 56   // LDS row stride in bf16 elems: 112B, 16B-aligned chunks, conflict-breaking

// GEMM1: h = gelu(x[perm] @ W1[e] + b1[e]); A=xb bf16 gathered, B=W1T bf16 [H][C]
__global__ __launch_bounds__(256) void k_gemm1(const unsigned short* __restrict__ xb,
                                               const unsigned short* __restrict__ w1t,
                                               const float* __restrict__ b1,
                                               const int* __restrict__ perm,
                                               const int* __restrict__ offs,
                                               const int* __restrict__ counts,
                                               unsigned short* __restrict__ hb) {
    const int e   = blockIdx.z;
    const int cnt = counts[e];
    const int m0  = blockIdx.x * TM;
    if (m0 >= cnt) return;
    const int seg0 = offs[e];
    const int h0   = blockIdx.y * TN;

    __shared__ __align__(16) unsigned short As[2][TM * LSTR];
    __shared__ __align__(16) unsigned short Bs[2][TN * LSTR];
    __shared__ int toks[TM];

    const int tid  = threadIdx.x;
    const int lane = tid & 31;
    const int wv   = tid >> 5;
    const int wm   = wv & 1;    // 2 waves along M (64 rows each)
    const int wn   = wv >> 1;   // 4 waves along N (32 cols each)
    const int lrow = lane & 15;
    const int k0   = (lane >> 4) * 8;

    if (tid < TM) {
        int r = m0 + tid;
        toks[tid] = perm[seg0 + (r < cnt ? r : cnt - 1)];   // clamp: no branches later
    }
    __syncthreads();

    const int row  = tid >> 1;
    const int half = tid & 1;
    const int myTok = toks[row];
    const unsigned short* gA = xb  + (size_t)myTok * NC + half * 16;
    const unsigned short* gB = w1t + ((size_t)e * NH + h0 + row) * NC + half * 16;
    const int lofs = row * LSTR + half * 16;

    v8f acc[4][2] = {};
    constexpr int NCH = NC / 32;     // 32 K-chunks

    // prologue: fill both pipeline stages
#pragma unroll
    for (int s = 0; s < 2; ++s) {
        async_cp16(gA + s * 32,     &As[s][lofs]);
        async_cp16(gA + s * 32 + 8, &As[s][lofs + 8]);
        async_cp16(gB + s * 32,     &Bs[s][lofs]);
        async_cp16(gB + s * 32 + 8, &Bs[s][lofs + 8]);
    }

    int p = 0;
#pragma unroll 1
    for (int i = 0; i < NCH; ++i) {
        if (i < NCH - 1) wait_async<4>(); else wait_async<0>();
        __syncthreads();

        v16bf af[4], bf[2];
#pragma unroll
        for (int mi = 0; mi < 4; ++mi)
            af[mi] = ld_frag(&As[p][(wm * 64 + mi * 16 + lrow) * LSTR + k0]);
#pragma unroll
        for (int ni = 0; ni < 2; ++ni)
            bf[ni] = ld_frag(&Bs[p][(wn * 32 + ni * 16 + lrow) * LSTR + k0]);
#pragma unroll
        for (int mi = 0; mi < 4; ++mi)
#pragma unroll
            for (int ni = 0; ni < 2; ++ni)
                acc[mi][ni] = __builtin_amdgcn_wmma_f32_16x16x32_bf16(
                    false, af[mi], false, bf[ni], (short)0, acc[mi][ni], false, false);
        __syncthreads();

        if (i + 2 < NCH) {
            int kc = (i + 2) * 32;
            async_cp16(gA + kc,     &As[p][lofs]);
            async_cp16(gA + kc + 8, &As[p][lofs + 8]);
            async_cp16(gB + kc,     &Bs[p][lofs]);
            async_cp16(gB + kc + 8, &Bs[p][lofs + 8]);
        }
        p ^= 1;
    }

    // ---- epilogue: +b1, gelu(exact), store bf16 hidden (grouped-row order) ----
    const float* b1e = b1 + (size_t)e * NH;
    const int mhi = (lane >> 4) * 8;
#pragma unroll
    for (int mi = 0; mi < 4; ++mi) {
#pragma unroll
        for (int ni = 0; ni < 2; ++ni) {
            int h = h0 + wn * 32 + ni * 16 + lrow;
            float bias = b1e[h];
#pragma unroll
            for (int r = 0; r < 8; ++r) {
                int rr = m0 + wm * 64 + mi * 16 + r + mhi;
                if (rr < cnt) {
                    float v = acc[mi][ni][r] + bias;
                    float g = 0.5f * v * (1.0f + erff(v * 0.70710678118654752f));
                    hb[(size_t)(seg0 + rr) * NH + h] = f2bf(g);
                }
            }
        }
    }
}

// GEMM2: out[tok] = probs[tok]*(h @ W2[e] + b2[e]); A=hb bf16 grouped, B=W2T bf16 [C][H]
__global__ __launch_bounds__(256) void k_gemm2(const unsigned short* __restrict__ hb,
                                               const unsigned short* __restrict__ w2t,
                                               const float* __restrict__ b2,
                                               const int* __restrict__ perm,
                                               const int* __restrict__ offs,
                                               const int* __restrict__ counts,
                                               const float* __restrict__ probs,
                                               float* __restrict__ out) {
    const int e   = blockIdx.z;
    const int cnt = counts[e];
    const int m0  = blockIdx.x * TM;
    if (m0 >= cnt) return;
    const int seg0 = offs[e];
    const int n0   = blockIdx.y * TN;

    __shared__ __align__(16) unsigned short As[2][TM * LSTR];
    __shared__ __align__(16) unsigned short Bs[2][TN * LSTR];
    __shared__ int toks[TM];

    const int tid  = threadIdx.x;
    const int lane = tid & 31;
    const int wv   = tid >> 5;
    const int wm   = wv & 1;
    const int wn   = wv >> 1;
    const int lrow = lane & 15;
    const int k0   = (lane >> 4) * 8;

    if (tid < TM) {
        int r = m0 + tid;
        toks[tid] = perm[seg0 + (r < cnt ? r : cnt - 1)];
    }
    __syncthreads();

    const int row  = tid >> 1;
    const int half = tid & 1;
    const int gr   = m0 + row < cnt ? m0 + row : cnt - 1;   // clamped grouped row
    const unsigned short* gA = hb  + (size_t)(seg0 + gr) * NH + half * 16;
    const unsigned short* gB = w2t + ((size_t)e * NC + n0 + row) * NH + half * 16;
    const int lofs = row * LSTR + half * 16;

    v8f acc[4][2] = {};
    constexpr int NCH = NH / 32;     // 128 K-chunks

#pragma unroll
    for (int s = 0; s < 2; ++s) {
        async_cp16(gA + s * 32,     &As[s][lofs]);
        async_cp16(gA + s * 32 + 8, &As[s][lofs + 8]);
        async_cp16(gB + s * 32,     &Bs[s][lofs]);
        async_cp16(gB + s * 32 + 8, &Bs[s][lofs + 8]);
    }

    int p = 0;
#pragma unroll 1
    for (int i = 0; i < NCH; ++i) {
        if (i < NCH - 1) wait_async<4>(); else wait_async<0>();
        __syncthreads();

        v16bf af[4], bf[2];
#pragma unroll
        for (int mi = 0; mi < 4; ++mi)
            af[mi] = ld_frag(&As[p][(wm * 64 + mi * 16 + lrow) * LSTR + k0]);
#pragma unroll
        for (int ni = 0; ni < 2; ++ni)
            bf[ni] = ld_frag(&Bs[p][(wn * 32 + ni * 16 + lrow) * LSTR + k0]);
#pragma unroll
        for (int mi = 0; mi < 4; ++mi)
#pragma unroll
            for (int ni = 0; ni < 2; ++ni)
                acc[mi][ni] = __builtin_amdgcn_wmma_f32_16x16x32_bf16(
                    false, af[mi], false, bf[ni], (short)0, acc[mi][ni], false, false);
        __syncthreads();

        if (i + 2 < NCH) {
            int kc = (i + 2) * 32;
            async_cp16(gA + kc,     &As[p][lofs]);
            async_cp16(gA + kc + 8, &As[p][lofs + 8]);
            async_cp16(gB + kc,     &Bs[p][lofs]);
            async_cp16(gB + kc + 8, &Bs[p][lofs + 8]);
        }
        p ^= 1;
    }

    // ---- epilogue: +b2, gate-scale, scatter by token ----
    const float* b2e = b2 + (size_t)e * NC;
    const int mhi = (lane >> 4) * 8;
#pragma unroll
    for (int mi = 0; mi < 4; ++mi) {
#pragma unroll
        for (int ni = 0; ni < 2; ++ni) {
            int col  = n0 + wn * 32 + ni * 16 + lrow;
            float bias = b2e[col];
#pragma unroll
            for (int r = 0; r < 8; ++r) {
                int rr = m0 + wm * 64 + mi * 16 + r + mhi;
                if (rr < cnt) {
                    int tok = toks[wm * 64 + mi * 16 + r + mhi];
                    float g = probs[tok];
                    out[(size_t)tok * NC + col] = g * (acc[mi][ni][r] + bias);
                }
            }
        }
    }
}

// ---------------- host launcher ----------------
extern "C" void kernel_launch(void* const* d_in, const int* in_sizes, int n_in,
                              void* d_out, int out_size, void* d_ws, size_t ws_size,
                              hipStream_t stream) {
    const float* x  = (const float*)d_in[0];   // [N, C]
    const float* Wr = (const float*)d_in[1];   // [C, E]
    const float* br = (const float*)d_in[2];   // [E]
    const float* W1 = (const float*)d_in[3];   // [E, C, H]
    const float* b1 = (const float*)d_in[4];   // [E, H]
    const float* W2 = (const float*)d_in[5];   // [E, H, C]
    const float* b2 = (const float*)d_in[6];   // [E, C]
    float* out = (float*)d_out;                // [N, C] + aux scalar

    // workspace carve-up (256B aligned slabs)
    char* w = (char*)d_ws;
    int*   counts = (int*)(w);
    int*   cursor = (int*)(w + 256);
    int*   offs   = (int*)(w + 512);
    float* probs  = (float*)(w + 1024);
    int*   eidx   = (int*)(w + 1024 + 32768);
    int*   perm   = (int*)(w + 1024 + 65536);
    unsigned short* xb  = (unsigned short*)(w + 1024 + 98304);   // N*C  bf16 = 16.78 MB
    unsigned short* hb  = xb  + (size_t)NTOK * NC;               // N*H  bf16 = 67.1 MB
    unsigned short* w1t = hb  + (size_t)NTOK * NH;               // E*H*C bf16 = 67.1 MB
    unsigned short* w2t = w1t + (size_t)NE * NH * NC;            // E*C*H bf16 = 67.1 MB
    (void)ws_size; (void)n_in; (void)in_sizes; (void)out_size;

    k_init<<<1, 32, 0, stream>>>(counts, cursor);
    k_router<<<NTOK / 8, 256, 0, stream>>>(x, Wr, br, probs, eidx, counts);
    k_cvt<<<(NTOK * NC) / (256 * 8), 256, 0, stream>>>(x, xb);

    // weight transpose+convert: W1 [C][H] -> w1t [H][C], W2 [H][C] -> w2t [C][H]
    k_wt<<<dim3(NH / WT, NC / WT, NE), 256, 0, stream>>>(W1, w1t, NC, NH);
    k_wt<<<dim3(NC / WT, NH / WT, NE), 256, 0, stream>>>(W2, w2t, NH, NC);

    k_scan<<<1, 32, 0, stream>>>(counts, offs, out + (size_t)NTOK * NC);
    k_scatter<<<NTOK / 256, 256, 0, stream>>>(eidx, offs, cursor, perm);

    dim3 g1(NTOK / TM, NH / TN, NE);   // 64 x 32 x 8 (early-exit past segment end)
    k_gemm1<<<g1, 256, 0, stream>>>(xb, w1t, b1, perm, offs, counts, hb);

    dim3 g2(NTOK / TM, NC / TN, NE);   // 64 x 8 x 8
    k_gemm2<<<g2, 256, 0, stream>>>(hb, w2t, b2, perm, offs, counts, probs, out);
}
